// SpatialGCN_45071386804520
// MI455X (gfx1250) — matrix operand order repaired and checked by reading, compiled, not run on hardware
//
#include <hip/hip_runtime.h>

// ---------------------------------------------------------------------------
// SpatialGCN on MI455X (gfx1250): bf16 WMMA pipeline, wave32, 8 waves/block.
// Round 3: peel proj weight loop (global_load instead of flat_load for A
// fragments, decoupling DScnt); prefetch residual x tile in out_kernel.
// ---------------------------------------------------------------------------

typedef __attribute__((ext_vector_type(16))) __bf16    bf16x16;
typedef __attribute__((ext_vector_type(8)))  float     f32x8;
typedef __attribute__((ext_vector_type(4)))  float     f32x4;
typedef __attribute__((ext_vector_type(4)))  unsigned  u32x4;
typedef __attribute__((ext_vector_type(2)))  unsigned  u32x2;

#define NPIX 16384   // H*W
#define BATCH 8
#define PLANE 256
#define INTER 128

// Native converts (v_cvt_pk_bf16_f32) instead of manual bit-twiddled RNE.
__device__ __forceinline__ unsigned short f2bf_u(float f) {
    return __builtin_bit_cast(unsigned short, (__bf16)f);
}

union Frag16 {                                  // 16 bf16 = 8 VGPRs
    bf16x16        v;
    __bf16         h[16];
    unsigned short u16[16];
    u32x4          q[2];
};

__device__ __forceinline__ f32x8 wmma_bf16(const Frag16& a, const Frag16& b, f32x8 c) {
    return __builtin_amdgcn_wmma_f32_16x16x32_bf16(
        /*neg_a=*/false, a.v, /*neg_b=*/false, b.v,
        /*c_mod=*/(short)0, c, /*reuse_a=*/false, /*reuse_b=*/false);
}

// A-fragment from fp32 row-major weights: lane<16 row m=l15 K{ks..ks+15};
// lane>=16 same row, K{ks+16..ks+31} (contiguous order, matches B).
__device__ __forceinline__ void load_a_frag_f32(Frag16& af, const float* rowp) {
    f32x4 a0 = *(const f32x4*)(rowp);
    f32x4 a1 = *(const f32x4*)(rowp + 4);
    f32x4 a2 = *(const f32x4*)(rowp + 8);
    f32x4 a3 = *(const f32x4*)(rowp + 12);
#pragma unroll
    for (int i = 0; i < 4; ++i) {
        af.h[i]      = (__bf16)a0[i];
        af.h[4 + i]  = (__bf16)a1[i];
        af.h[8 + i]  = (__bf16)a2[i];
        af.h[12 + i] = (__bf16)a3[i];
    }
}

// Swizzled LDS index for [col][k] layouts (16-element chunk XOR): keeps
// 32B-aligned fragment reads while spreading banks.
__device__ __forceinline__ int sw256(int col, int k) {
    return col * 256 + ((((k >> 4) ^ col) & 15) << 4) + (k & 15);
}
__device__ __forceinline__ int sw128(int col, int k) {
    return col * 128 + ((((k >> 4) ^ col) & 7) << 4) + (k & 15);
}

// ---------------------------------------------------------------------------
__global__ void zero_kernel(float* __restrict__ p, int n) {
    int i = blockIdx.x * 256 + threadIdx.x;
    if (i < n) p[i] = 0.f;
}

// ---------------------------------------------------------------------------
// Kernel 1: n_j = w_j @ x + b_j for j=1..3.  [128x256]x[256x128]-tile GEMM.
// Block = (batch, 128-pixel tile). x tile staged fp32->bf16 into LDS as [n][k].
// One inlined instance per weight so each A-load keeps a direct kernarg base
// (global_load_b128, LOADcnt only — no flat/DScnt coupling).
// ---------------------------------------------------------------------------
__device__ __forceinline__ void proj_one(
    const unsigned short* xl, const float* __restrict__ W,
    const float* __restrict__ Bv, unsigned short* __restrict__ O,
    int mrow, int l15, int lh, int n0)
{
    f32x8 acc[8] = {};
#pragma unroll 1
    for (int ks = 0; ks < PLANE; ks += 32) {
        Frag16 af;
        load_a_frag_f32(af, W + (mrow + l15) * PLANE + ks + lh * 16);
        const int kb = ks + lh * 16;           // B: lane holds col, 16 contig K
        Frag16 bf_[8];
#pragma unroll
        for (int tn = 0; tn < 8; ++tn) {
            const u32x4* p = (const u32x4*)&xl[sw256(tn * 16 + l15, kb)];
            bf_[tn].q[0] = p[0]; bf_[tn].q[1] = p[1];
        }
#pragma unroll
        for (int tn = 0; tn < 8; ++tn)
            acc[tn] = wmma_bf16(af, bf_[tn], acc[tn]);
    }
#pragma unroll
    for (int tn = 0; tn < 8; ++tn) {
        const int col = n0 + tn * 16 + l15;
#pragma unroll
        for (int r = 0; r < 8; ++r) {
            const int row = mrow + r + lh * 8;
            O[(size_t)row * NPIX + col] = f2bf_u(acc[tn][r] + Bv[row]);
        }
    }
}

__global__ __launch_bounds__(256) void proj_kernel(
    const float* __restrict__ x,
    const float* __restrict__ w1, const float* __restrict__ b1,
    const float* __restrict__ w2, const float* __restrict__ b2,
    const float* __restrict__ w3, const float* __restrict__ b3,
    unsigned short* __restrict__ o1, unsigned short* __restrict__ o2,
    unsigned short* __restrict__ o3)
{
    __shared__ unsigned short xl[128 * 256];   // 64 KB, swizzled [n][k]
    const int b  = blockIdx.y;
    const int n0 = blockIdx.x * 128;
    const int t = threadIdx.x;
    const int lane = t & 31, wv = t >> 5;
    const int l15 = lane & 15, lh = lane >> 4;

    const float* xb = x + (size_t)b * PLANE * NPIX;
#pragma unroll 4
    for (int it = 0; it < 32; ++it) {          // 256 k-rows, 8 rows/iter
        const int k  = it * 8 + wv;
        const int nc = lane * 4;
        f32x4 v = *(const f32x4*)(xb + (size_t)k * NPIX + n0 + nc);
#pragma unroll
        for (int i = 0; i < 4; ++i) xl[sw256(nc + i, k)] = f2bf_u(v[i]);
    }
    __syncthreads();

    const int mrow = wv * 16;                  // wave's 16-row output strip
    const size_t ob = (size_t)b * INTER * NPIX;
    proj_one(xl, w1, b1, o1 + ob, mrow, l15, lh, n0);
    proj_one(xl, w2, b2, o2 + ob, mrow, l15, lh, n0);
    proj_one(xl, w3, b3, o3 + ob, mrow, l15, lh, n0);
}

// ---------------------------------------------------------------------------
// Kernel 2: aff[c][d] += sum_n n3[c][n]*n2[d][n].  Split-K (32 chunks of 512)
// with fp32 global atomics. Both operands have contiguous K (=n) in memory,
// so fragments load straight from global (L2-resident reuse).
// ---------------------------------------------------------------------------
__global__ __launch_bounds__(256) void aff_kernel(
    const unsigned short* __restrict__ n3,
    const unsigned short* __restrict__ n2,
    float* __restrict__ aff)
{
    const int b  = blockIdx.y;
    const int k0 = blockIdx.x * 512;
    const int t = threadIdx.x;
    const int lane = t & 31, wv = t >> 5;
    const int l15 = lane & 15, lh = lane >> 4;

    const unsigned short* A  = n3 + (size_t)b * INTER * NPIX
                                  + (size_t)(wv * 16 + l15) * NPIX;
    const unsigned short* Bb = n2 + (size_t)b * INTER * NPIX
                                  + (size_t)l15 * NPIX;

    f32x8 acc[8] = {};
#pragma unroll 1
    for (int ks = k0; ks < k0 + 512; ks += 32) {
        Frag16 af;
        const unsigned short* ap = A + ks + lh * 16;
        af.q[0] = *(const u32x4*)(ap);
        af.q[1] = *(const u32x4*)(ap + 8);
        const int kb = ks + lh * 16;
        Frag16 bf_[8];                         // preload: global_load burst
#pragma unroll
        for (int tn = 0; tn < 8; ++tn) {
            const unsigned short* bp = Bb + (size_t)(tn * 16) * NPIX + kb;
            bf_[tn].q[0] = *(const u32x4*)(bp);
            bf_[tn].q[1] = *(const u32x4*)(bp + 8);
        }
#pragma unroll
        for (int tn = 0; tn < 8; ++tn)
            acc[tn] = wmma_bf16(af, bf_[tn], acc[tn]);
    }
    float* AF = aff + (size_t)b * INTER * INTER;
#pragma unroll
    for (int tn = 0; tn < 8; ++tn) {
        const int d = tn * 16 + l15;
#pragma unroll
        for (int r = 0; r < 8; ++r) {
            const int c = wv * 16 + r + lh * 8;
            atomicAdd(AF + c * INTER + d, acc[tn][r]);
        }
    }
}

// ---------------------------------------------------------------------------
// Kernel 3: row softmax over d, store transposed bf16 afT[d][c].
// One wave32 per row; shuffle reductions.
// ---------------------------------------------------------------------------
__global__ __launch_bounds__(256) void softmax_kernel(
    const float* __restrict__ aff, unsigned short* __restrict__ afT)
{
    const int gw = blockIdx.x * 8 + (threadIdx.x >> 5);  // global wave = row
    const int b = gw >> 7, c = gw & 127;
    const int lane = threadIdx.x & 31;
    const float* row = aff + ((size_t)b * INTER + c) * INTER;

    f32x4 v = *(const f32x4*)(row + lane * 4);
    float m = fmaxf(fmaxf(v[0], v[1]), fmaxf(v[2], v[3]));
#pragma unroll
    for (int off = 16; off > 0; off >>= 1) m = fmaxf(m, __shfl_xor(m, off, 32));
    float s = 0.f;
#pragma unroll
    for (int i = 0; i < 4; ++i) { v[i] = __expf(v[i] - m); s += v[i]; }
#pragma unroll
    for (int off = 16; off > 0; off >>= 1) s += __shfl_xor(s, off, 32);
    const float inv = 1.f / s;
    unsigned short* T = afT + (size_t)b * INTER * INTER;
#pragma unroll
    for (int i = 0; i < 4; ++i) T[(lane * 4 + i) * INTER + c] = f2bf_u(v[i] * inv);
}

// ---------------------------------------------------------------------------
// Kernel 4 (fused): av = afT @ n1  ->  avw = wg @ av  ->  BN affine -> bf16.
// n1 tile staged to LDS [n][c]; av tile bounced through LDS [n][d].
// ---------------------------------------------------------------------------
__global__ __launch_bounds__(256) void avw_kernel(
    const unsigned short* __restrict__ n1,
    const unsigned short* __restrict__ afT,
    const float* __restrict__ wg,
    const float* __restrict__ gamma, const float* __restrict__ beta,
    const float* __restrict__ rmean, const float* __restrict__ rvar,
    unsigned short* __restrict__ avw)
{
    __shared__ unsigned short s1[128 * 128];   // 32 KB [n][c]
    __shared__ unsigned short s2[128 * 128];   // 32 KB [n][d]
    const int b  = blockIdx.y;
    const int n0 = blockIdx.x * 128;
    const int t = threadIdx.x;
    const int lane = t & 31, wv = t >> 5;
    const int l15 = lane & 15, lh = lane >> 4;

    const unsigned short* N1 = n1 + (size_t)b * INTER * NPIX;
#pragma unroll 2
    for (int it = 0; it < 16; ++it) {
        const int c  = it * 8 + wv;
        const int nc = lane * 4;
        u32x2 dv = *(const u32x2*)(N1 + (size_t)c * NPIX + n0 + nc);
        s1[sw128(nc + 0, c)] = (unsigned short)(dv[0] & 0xffffu);
        s1[sw128(nc + 1, c)] = (unsigned short)(dv[0] >> 16);
        s1[sw128(nc + 2, c)] = (unsigned short)(dv[1] & 0xffffu);
        s1[sw128(nc + 3, c)] = (unsigned short)(dv[1] >> 16);
    }
    __syncthreads();

    {   // Stage A: av[d][n] = sum_c afT[d][c] * n1[c][n]
        f32x8 acc[8] = {};
        const unsigned short* AT = afT + (size_t)b * INTER * INTER
                                       + (wv * 16 + l15) * INTER;
#pragma unroll 1
        for (int ks = 0; ks < INTER; ks += 32) {
            Frag16 af;
            af.q[0] = *(const u32x4*)(AT + ks + lh * 16);
            af.q[1] = *(const u32x4*)(AT + ks + lh * 16 + 8);
            const int kb = ks + lh * 16;
            Frag16 bf_[8];
#pragma unroll
            for (int tn = 0; tn < 8; ++tn) {
                const u32x4* p = (const u32x4*)&s1[sw128(tn * 16 + l15, kb)];
                bf_[tn].q[0] = p[0]; bf_[tn].q[1] = p[1];
            }
#pragma unroll
            for (int tn = 0; tn < 8; ++tn)
                acc[tn] = wmma_bf16(af, bf_[tn], acc[tn]);
        }
#pragma unroll
        for (int tn = 0; tn < 8; ++tn) {
            const int col = tn * 16 + l15;
#pragma unroll
            for (int r = 0; r < 8; ++r) {
                const int d = wv * 16 + r + lh * 8;
                s2[sw128(col, d)] = f2bf_u(acc[tn][r]);
            }
        }
    }
    __syncthreads();

    {   // Stage B: avw[e][n] = sum_d wg[e][d] * av[d][n], then BN affine
        f32x8 acc[8] = {};
        const int erow = wv * 16;
#pragma unroll 1
        for (int ks = 0; ks < INTER; ks += 32) {
            Frag16 af;
            load_a_frag_f32(af, wg + (erow + l15) * INTER + ks + lh * 16);
            const int kb = ks + lh * 16;
            Frag16 bf_[8];
#pragma unroll
            for (int tn = 0; tn < 8; ++tn) {
                const u32x4* p = (const u32x4*)&s2[sw128(tn * 16 + l15, kb)];
                bf_[tn].q[0] = p[0]; bf_[tn].q[1] = p[1];
            }
#pragma unroll
            for (int tn = 0; tn < 8; ++tn)
                acc[tn] = wmma_bf16(af, bf_[tn], acc[tn]);
        }
        unsigned short* O = avw + (size_t)b * INTER * NPIX;
#pragma unroll
        for (int r = 0; r < 8; ++r) {
            const int e = erow + r + lh * 8;
            const float iv = gamma[e] * rsqrtf(rvar[e] + 1e-5f);
            const float ad = beta[e] - rmean[e] * iv;
#pragma unroll
            for (int tn = 0; tn < 8; ++tn) {
                const int col = n0 + tn * 16 + l15;
                O[(size_t)e * NPIX + col] = f2bf_u(acc[tn][r] * iv + ad);
            }
        }
    }
}

// ---------------------------------------------------------------------------
// Kernel 5: o = relu(wo @ avw + bo + x).  Each wave covers 32 output rows
// (two 16-row strips) so 8 waves cover all 256 PLANE channels.  Residual x
// tile is prefetched (global_prefetch_b8) before the WMMA phase.
// ---------------------------------------------------------------------------
__global__ __launch_bounds__(256) void out_kernel(
    const unsigned short* __restrict__ avw,
    const float* __restrict__ wo, const float* __restrict__ bo,
    const float* __restrict__ x, float* __restrict__ out)
{
    __shared__ unsigned short sv[128 * 128];   // 32 KB [n][e]
    const int b  = blockIdx.y;
    const int n0 = blockIdx.x * 128;
    const int t = threadIdx.x;
    const int lane = t & 31, wv = t >> 5;
    const int l15 = lane & 15, lh = lane >> 4;

    const float* xb = x + (size_t)b * PLANE * NPIX;
    // Prefetch residual tile (256 rows x 512B) so epilogue reads hit cache.
#pragma unroll 4
    for (int it = 0; it < 32; ++it) {
        const int row = it * 8 + wv;
        __builtin_prefetch(xb + (size_t)row * NPIX + n0 + lane * 4, 0, 0);
    }

    const unsigned short* AV = avw + (size_t)b * INTER * NPIX;
#pragma unroll 2
    for (int it = 0; it < 16; ++it) {
        const int e  = it * 8 + wv;
        const int nc = lane * 4;
        u32x2 dv = *(const u32x2*)(AV + (size_t)e * NPIX + n0 + nc);
        sv[sw128(nc + 0, e)] = (unsigned short)(dv[0] & 0xffffu);
        sv[sw128(nc + 1, e)] = (unsigned short)(dv[0] >> 16);
        sv[sw128(nc + 2, e)] = (unsigned short)(dv[1] & 0xffffu);
        sv[sw128(nc + 3, e)] = (unsigned short)(dv[1] >> 16);
    }
    __syncthreads();

    float* ob = out + (size_t)b * PLANE * NPIX;
#pragma unroll 1
    for (int half = 0; half < 2; ++half) {
        const int mrow = wv * 32 + half * 16;
        f32x8 acc[8] = {};
#pragma unroll 1
        for (int ks = 0; ks < INTER; ks += 32) {
            Frag16 af;
            load_a_frag_f32(af, wo + (mrow + l15) * INTER + ks + lh * 16);
            const int kb = ks + lh * 16;
            Frag16 bf_[8];
#pragma unroll
            for (int tn = 0; tn < 8; ++tn) {
                const u32x4* p = (const u32x4*)&sv[sw128(tn * 16 + l15, kb)];
                bf_[tn].q[0] = p[0]; bf_[tn].q[1] = p[1];
            }
#pragma unroll
            for (int tn = 0; tn < 8; ++tn)
                acc[tn] = wmma_bf16(af, bf_[tn], acc[tn]);
        }
#pragma unroll
        for (int tn = 0; tn < 8; ++tn) {
            const int col = n0 + tn * 16 + l15;
#pragma unroll
            for (int r = 0; r < 8; ++r) {
                const int row = mrow + r + lh * 8;
                const size_t idx = (size_t)row * NPIX + col;
                ob[idx] = fmaxf(acc[tn][r] + bo[row] + xb[idx], 0.f);
            }
        }
    }
}

// ---------------------------------------------------------------------------
extern "C" void kernel_launch(void* const* d_in, const int* in_sizes, int n_in,
                              void* d_out, int out_size, void* d_ws, size_t ws_size,
                              hipStream_t stream) {
    (void)in_sizes; (void)n_in; (void)out_size; (void)ws_size;
    const float* x     = (const float*)d_in[0];
    const float* w1    = (const float*)d_in[1];
    const float* b1    = (const float*)d_in[2];
    const float* w2    = (const float*)d_in[3];
    const float* b2    = (const float*)d_in[4];
    const float* w3    = (const float*)d_in[5];
    const float* b3    = (const float*)d_in[6];
    const float* wg    = (const float*)d_in[7];
    const float* gamma = (const float*)d_in[8];
    const float* beta  = (const float*)d_in[9];
    const float* rmean = (const float*)d_in[10];
    const float* rvar  = (const float*)d_in[11];
    const float* wo    = (const float*)d_in[12];
    const float* bo    = (const float*)d_in[13];
    float* out = (float*)d_out;

    // Workspace layout (bf16 intermediates): 4 x 32MB + aff fp32 + afT bf16
    char* ws = (char*)d_ws;
    const size_t szn = (size_t)BATCH * INTER * NPIX * sizeof(unsigned short);
    unsigned short* n1  = (unsigned short*)(ws);
    unsigned short* n2  = (unsigned short*)(ws + szn);
    unsigned short* n3  = (unsigned short*)(ws + 2 * szn);
    unsigned short* avw = (unsigned short*)(ws + 3 * szn);
    float*          aff = (float*)(ws + 4 * szn);
    unsigned short* afT = (unsigned short*)(ws + 4 * szn
                                            + (size_t)BATCH * INTER * INTER * 4);

    zero_kernel<<<(BATCH * INTER * INTER + 255) / 256, 256, 0, stream>>>(
        aff, BATCH * INTER * INTER);
    proj_kernel<<<dim3(NPIX / 128, BATCH), 256, 0, stream>>>(
        x, w1, b1, w2, b2, w3, b3, n1, n2, n3);
    aff_kernel<<<dim3(NPIX / 512, BATCH), 256, 0, stream>>>(n3, n2, aff);
    softmax_kernel<<<(BATCH * INTER) / 8, 256, 0, stream>>>(aff, afT);
    avw_kernel<<<dim3(NPIX / 128, BATCH), 256, 0, stream>>>(
        n1, afT, wg, gamma, beta, rmean, rvar, avw);
    out_kernel<<<dim3(NPIX / 128, BATCH), 256, 0, stream>>>(avw, wo, bo, x, out);
}
